// DifferentialAttention_51161650430156
// MI455X (gfx1250) — matrix operand order repaired and checked by reading, not compiled, not run on hardware
//
#include <hip/hip_runtime.h>

typedef __attribute__((ext_vector_type(16))) __bf16 v16bf;
typedef __attribute__((ext_vector_type(8)))  __bf16 v8bf;
typedef __attribute__((ext_vector_type(8)))  float  v8f;

#define WMMA_BF16(a, b, c) \
  __builtin_amdgcn_wmma_f32_16x16x32_bf16(false, (a), false, (b), (short)0, (c), false, false)

constexpr int Bc = 4;      // batch
constexpr int Sc = 4096;   // sequence
constexpr int Dc = 256;    // model dim
constexpr int Kc = 64;     // key dim

// ---------------------------------------------------------------------------
// Kernel 0: convert + transpose weights  W[k][n] (f32) -> Wt[mat][n][k] (bf16)
// ---------------------------------------------------------------------------
__global__ __launch_bounds__(256) void wt_convert(const float* __restrict__ Wq,
                                                  const float* __restrict__ Wk,
                                                  const float* __restrict__ Wv,
                                                  __bf16* __restrict__ Wt) {
  int idx = blockIdx.x * 256 + threadIdx.x;
  if (idx >= 3 * Kc * Dc) return;
  int mat = idx / (Kc * Dc);
  int rem = idx % (Kc * Dc);
  int n = rem / Dc;
  int k = rem % Dc;
  const float* W = (mat == 0) ? Wq : (mat == 1) ? Wk : Wv;
  Wt[idx] = (__bf16)W[k * Kc + n];
}

// ---------------------------------------------------------------------------
// Kernel 1: fused QKV projection (bf16 WMMA, f32 accum, bias add)
//   Q,K stored row-major [B,S,64] bf16;  V stored transposed [B,64,S] bf16
// ---------------------------------------------------------------------------
__global__ __launch_bounds__(128) void qkv_proj(const float* __restrict__ x,
                                                const __bf16* __restrict__ Wt,
                                                const float* __restrict__ bq,
                                                const float* __restrict__ bk,
                                                const float* __restrict__ bv,
                                                __bf16* __restrict__ Q,
                                                __bf16* __restrict__ K,
                                                __bf16* __restrict__ Vt) {
  constexpr int XP = Dc + 8;  // padded LDS row stride (bf16) -> conflict-free b128
  __shared__ __bf16 xs[16 * XP];

  const int b    = blockIdx.x >> 8;
  const int row0 = (blockIdx.x & 255) << 4;
  const float* xg = x + ((size_t)b * Sc + row0) * Dc;
  for (int i = threadIdx.x; i < 16 * Dc; i += 128)
    xs[(i >> 8) * XP + (i & (Dc - 1))] = (__bf16)xg[i];
  __syncthreads();

  const int lane = threadIdx.x & 31;
  const int wave = threadIdx.x >> 5;
  const int lr   = lane & 15;
  const int half = lane >> 4;
  const int n    = wave * 16 + lr;  // output column this lane owns

  v8f accQ, accK, accV;
  const float vbq = bq[n], vbk = bk[n], vbv = bv[n];
#pragma unroll
  for (int r = 0; r < 8; ++r) { accQ[r] = vbq; accK[r] = vbk; accV[r] = vbv; }

  const __bf16* xrow = xs + lr * XP;
  const __bf16* wqp  = Wt + (size_t)(0 * Kc + n) * Dc + 16 * half;
  const __bf16* wkp  = Wt + (size_t)(1 * Kc + n) * Dc + 16 * half;
  const __bf16* wvp  = Wt + (size_t)(2 * Kc + n) * Dc + 16 * half;

#pragma unroll
  for (int kk = 0; kk < Dc; kk += 32) {
    v8bf c0 = *(const v8bf*)(xrow + kk + 16 * half);
    v8bf c1 = *(const v8bf*)(xrow + kk + 16 * half + 8);
    v16bf a = __builtin_shufflevector(c0, c1, 0, 1, 2, 3, 4, 5, 6, 7,
                                      8, 9, 10, 11, 12, 13, 14, 15);
    v16bf bq_ = *(const v16bf*)(wqp + kk);
    v16bf bk_ = *(const v16bf*)(wkp + kk);
    v16bf bv_ = *(const v16bf*)(wvp + kk);
    accQ = WMMA_BF16(a, bq_, accQ);
    accK = WMMA_BF16(a, bk_, accK);
    accV = WMMA_BF16(a, bv_, accV);
  }

  __bf16* Qg = Q + ((size_t)b * Sc + row0) * Kc;
  __bf16* Kg = K + ((size_t)b * Sc + row0) * Kc;
  __bf16* Vg = Vt + ((size_t)b * Kc + n) * Sc + row0;
#pragma unroll
  for (int r = 0; r < 8; ++r) {
    int row = r + 8 * half;
    Qg[row * Kc + n] = (__bf16)accQ[r];
    Kg[row * Kc + n] = (__bf16)accK[r];
    Vg[row]          = (__bf16)accV[r];
  }
}

// ---------------------------------------------------------------------------
// Kernel 2: attention for one 16-row query slab, scores resident in LDS.
//   mean(softmax row) == 1/S exactly, so mask is  e_ij > Z_i / S.
// ---------------------------------------------------------------------------
constexpr int SROW = Sc + 4;  // padded f32 stride: 4100 % 64 == 4 -> conflict-free
constexpr int QP   = Kc + 8;  // padded bf16 stride: 72

constexpr size_t LDS_QS    = (size_t)16 * SROW * 4;        // scores: 262400 B
constexpr size_t LDS_RED   = LDS_QS + (size_t)16 * QP * 2; // +2304  = 264704
constexpr size_t LDS_MROW  = LDS_RED + (size_t)8 * 1024 * 4;  // +32768 = 297472
constexpr size_t LDS_ZROW  = LDS_MROW + 64;
constexpr size_t LDS_TOTAL = LDS_ZROW + 64;                // 297600 B < 320 KB

__global__ __launch_bounds__(256) void attn(const __bf16* __restrict__ Q,
                                            const __bf16* __restrict__ K,
                                            const __bf16* __restrict__ Vt,
                                            float* __restrict__ out) {
  extern __shared__ char smraw[];
  float*  sc   = (float*)(smraw);
  __bf16* qs   = (__bf16*)(smraw + LDS_QS);
  float*  red  = (float*)(smraw + LDS_RED);
  float*  mrow = (float*)(smraw + LDS_MROW);
  float*  zrow = (float*)(smraw + LDS_ZROW);

  const int b    = blockIdx.x >> 8;
  const int row0 = (blockIdx.x & 255) << 4;
  const __bf16* Qg = Q + ((size_t)b * Sc + row0) * Kc;
  const __bf16* Kg = K + (size_t)b * Sc * Kc;
  const __bf16* Vb = Vt + (size_t)b * Kc * Sc;

  for (int i = threadIdx.x; i < 16 * Kc; i += 256)
    qs[(i >> 6) * QP + (i & 63)] = Qg[i];
  __syncthreads();

  const int lane = threadIdx.x & 31;
  const int wave = threadIdx.x >> 5;
  const int lr   = lane & 15;
  const int half = lane >> 4;

  // ---- Phase 1: scores = Q * K^T  (each wave: 32 column tiles of 16) ----
  const __bf16* qrow = qs + lr * QP;
  for (int t = 0; t < 32; ++t) {
    const int nbase = (wave * 32 + t) * 16;
    v8f acc = {0.f, 0.f, 0.f, 0.f, 0.f, 0.f, 0.f, 0.f};
    const __bf16* krow = Kg + (size_t)(nbase + lr) * Kc + 16 * half;
#pragma unroll
    for (int kk = 0; kk < Kc; kk += 32) {
      v8bf c0 = *(const v8bf*)(qrow + kk + 16 * half);
      v8bf c1 = *(const v8bf*)(qrow + kk + 16 * half + 8);
      v16bf a = __builtin_shufflevector(c0, c1, 0, 1, 2, 3, 4, 5, 6, 7,
                                        8, 9, 10, 11, 12, 13, 14, 15);
      v16bf bb = *(const v16bf*)(krow + kk);
      acc = WMMA_BF16(a, bb, acc);
    }
#pragma unroll
    for (int r = 0; r < 8; ++r)
      sc[(r + 8 * half) * SROW + nbase + lr] = acc[r];
  }
  __syncthreads();

  // ---- Phase 2: row max, exp in place, row sum Z ----
  const int tr = threadIdx.x >> 4;  // row 0..15
  const int tc = threadIdx.x & 15;
  float* srow = sc + tr * SROW;
  float mx = -3.402823466e38f;
  for (int c = tc; c < Sc; c += 16) mx = fmaxf(mx, srow[c]);
  red[threadIdx.x] = mx;
  __syncthreads();
  if (tc == 0) {
    float m = red[tr * 16];
    for (int i = 1; i < 16; ++i) m = fmaxf(m, red[tr * 16 + i]);
    mrow[tr] = m;
  }
  __syncthreads();
  const float m = mrow[tr];
  float zsum = 0.f;
  for (int c = tc; c < Sc; c += 16) {
    float e = __expf(srow[c] - m);
    srow[c] = e;
    zsum += e;
  }
  red[threadIdx.x] = zsum;
  __syncthreads();
  if (tc == 0) {
    float z = 0.f;
    for (int i = 0; i < 16; ++i) z += red[tr * 16 + i];
    zrow[tr] = z;
  }
  __syncthreads();

  // ---- Phase 3: out = (masked P) @ V, split-K across 8 waves ----
  const float th = zrow[lr] * (1.0f / (float)Sc);  // mask: e > Z/S
  v8f o0 = {0.f, 0.f, 0.f, 0.f, 0.f, 0.f, 0.f, 0.f};
  v8f o1 = o0, o2 = o0, o3 = o0;
  const float* prow = sc + lr * SROW;
  const int kbeg = wave * (Sc / 8);
  for (int it = 0; it < (Sc / 8) / 32; ++it) {
    const int kk = kbeg + it * 32;
    v16bf a;
#pragma unroll
    for (int j = 0; j < 8; ++j) {
      float v0 = prow[kk + 16 * half + j];
      float v1 = prow[kk + 16 * half + 8 + j];
      a[j]     = (__bf16)(v0 > th ? v0 : 0.f);
      a[8 + j] = (__bf16)(v1 > th ? v1 : 0.f);
    }
    const __bf16* vbase = Vb + (size_t)lr * Sc + kk + 16 * half;
    v16bf b0 = *(const v16bf*)(vbase);
    v16bf b1 = *(const v16bf*)(vbase + (size_t)16 * Sc);
    v16bf b2 = *(const v16bf*)(vbase + (size_t)32 * Sc);
    v16bf b3 = *(const v16bf*)(vbase + (size_t)48 * Sc);
    o0 = WMMA_BF16(a, b0, o0);
    o1 = WMMA_BF16(a, b1, o1);
    o2 = WMMA_BF16(a, b2, o2);
    o3 = WMMA_BF16(a, b3, o3);
  }
  float* rw = red + wave * 1024;
#pragma unroll
  for (int r = 0; r < 8; ++r) {
    int row = r + 8 * half;
    rw[row * 64 + lr]      = o0[r];
    rw[row * 64 + 16 + lr] = o1[r];
    rw[row * 64 + 32 + lr] = o2[r];
    rw[row * 64 + 48 + lr] = o3[r];
  }
  __syncthreads();

  float* og = out + ((size_t)b * Sc + row0) * Kc;
  for (int o = threadIdx.x; o < 16 * Kc; o += 256) {
    float s = 0.f;
#pragma unroll
    for (int w = 0; w < 8; ++w) s += red[w * 1024 + o];
    og[o] = s / zrow[o >> 6];
  }
}

// ---------------------------------------------------------------------------
extern "C" void kernel_launch(void* const* d_in, const int* in_sizes, int n_in,
                              void* d_out, int out_size, void* d_ws, size_t ws_size,
                              hipStream_t stream) {
  const float* x  = (const float*)d_in[0];
  const float* Wq = (const float*)d_in[1];
  const float* bq = (const float*)d_in[2];
  const float* Wk = (const float*)d_in[3];
  const float* bk = (const float*)d_in[4];
  const float* Wv = (const float*)d_in[5];
  const float* bv = (const float*)d_in[6];
  float* out = (float*)d_out;

  char* ws = (char*)d_ws;
  __bf16* Qbf = (__bf16*)(ws + (size_t)0);
  __bf16* Kbf = (__bf16*)(ws + (size_t)2 * 1024 * 1024);
  __bf16* Vt  = (__bf16*)(ws + (size_t)4 * 1024 * 1024);
  __bf16* Wt  = (__bf16*)(ws + (size_t)6 * 1024 * 1024);

  wt_convert<<<(3 * Kc * Dc + 255) / 256, 256, 0, stream>>>(Wq, Wk, Wv, Wt);
  qkv_proj<<<Bc * (Sc / 16), 128, 0, stream>>>(x, Wt, bq, bk, bv, Qbf, Kbf, Vt);
  attn<<<Bc * (Sc / 16), 256, LDS_TOTAL, stream>>>(Qbf, Kbf, Vt, out);
}
